// DIT_7387343749473
// MI455X (gfx1250) — matrix operand order repaired and checked by reading, not compiled, not run on hardware
//
#include <hip/hip_runtime.h>

// ---------------------------------------------------------------------------
// DiT forward for gfx1250 (MI455X). All GEMMs + attention use
// v_wmma_f32_16x16x32_bf16 (bf16 in, f32 accumulate). GEMM waves compute a
// 64x32 register-blocked tile (8 wmma / iter, 4x weight reuse). LayerNorm,
// softmax, RoPE, SiLU/GELU epilogues in f32. Vectorized b128 fragment loads.
// ---------------------------------------------------------------------------

typedef __bf16 v16bf __attribute__((ext_vector_type(16)));
typedef __bf16 bf8v  __attribute__((ext_vector_type(8)));
typedef float  v8f   __attribute__((ext_vector_type(8)));
typedef float  f4v   __attribute__((ext_vector_type(4)));

#define Ln   4
#define Dm   768
#define Hn   12
#define HDn  64
#define CONDn 128
#define FREQn 256
#define Vn   32000
#define Bn   2
#define Sn   1024
#define MLPn 3072
#define Rn   (Bn*Sn)       // 2048 rows
#define LOG1E4 9.210340371976184f

#define SHUF16(lo, hi) __builtin_shufflevector(lo, hi, 0,1,2,3,4,5,6,7,8,9,10,11,12,13,14,15)

static __device__ __forceinline__ __bf16 f2bf(float f) { return (__bf16)f; }
static __device__ __forceinline__ float siluf(float x) { return x / (1.0f + __expf(-x)); }
static __device__ __forceinline__ float geluf(float x) {
  return 0.5f * x * (1.0f + tanhf(0.7978845608028654f * (x + 0.044715f * x * x * x)));
}

// --------------------------- tiny conditioning kernels ----------------------

__global__ void k_embed_gather(const int* __restrict__ idx, const float* __restrict__ emb,
                               float* __restrict__ x) {
  int i = blockIdx.x * 256 + threadIdx.x;           // Rn*Dm threads
  int row = i / Dm, j = i - row * Dm;
  x[i] = emb[(size_t)idx[row] * Dm + j];
}

__global__ void k_t1(const float* __restrict__ sigma, const float* __restrict__ w1,
                     const float* __restrict__ b1, float* __restrict__ t1) {
  int b = blockIdx.x, i = threadIdx.x;              // Bn blocks x CONDn threads
  float sg = sigma[b];
  float acc = b1[i];
  const float* wr = w1 + (size_t)i * FREQn;
  for (int j = 0; j < FREQn / 2; ++j) {
    float fj = __expf(-LOG1E4 * (float)j / (FREQn / 2));
    float t = sg * fj;
    acc += cosf(t) * wr[j] + sinf(t) * wr[j + FREQn / 2];
  }
  t1[b * CONDn + i] = siluf(acc);
}

__global__ void k_t2(const float* __restrict__ t1, const float* __restrict__ w2,
                     const float* __restrict__ b2, float* __restrict__ c) {
  int b = blockIdx.x, i = threadIdx.x;
  float acc = b2[i];
  const float* wr = w2 + (size_t)i * CONDn;
  for (int j = 0; j < CONDn; ++j) acc += t1[b * CONDn + j] * wr[j];
  c[b * CONDn + i] = siluf(acc);
}

__global__ void k_c6(const float* __restrict__ c, const float* __restrict__ aw,
                     const float* __restrict__ ab, float* __restrict__ c6) {
  int i = blockIdx.x * 256 + threadIdx.x;           // Ln*Bn*6D threads
  int j = i % (6 * Dm);
  int b = (i / (6 * Dm)) % Bn;
  int l = i / (Bn * 6 * Dm);
  float acc = ab[(size_t)l * 6 * Dm + j];
  const float* wr = aw + ((size_t)l * 6 * Dm + j) * CONDn;
  for (int k = 0; k < CONDn; ++k) acc += c[b * CONDn + k] * wr[k];
  c6[((size_t)l * Bn + b) * 6 * Dm + j] = acc;
}

__global__ void k_c2(const float* __restrict__ c, const float* __restrict__ aw,
                     const float* __restrict__ ab, float* __restrict__ c2) {
  int i = blockIdx.x * 256 + threadIdx.x;           // Bn*2D threads
  int j = i % (2 * Dm);
  int b = i / (2 * Dm);
  float acc = ab[j];
  const float* wr = aw + (size_t)j * CONDn;
  for (int k = 0; k < CONDn; ++k) acc += c[b * CONDn + k] * wr[k];
  c2[(size_t)b * 2 * Dm + j] = acc;
}

// --------------------------- LayerNorm + adaLN modulate ---------------------
__global__ void __launch_bounds__(256)
k_ln_mod(const float* __restrict__ x, const float* __restrict__ w,
         const float* __restrict__ cm, int cstride, int sh_off, int sc_off,
         __bf16* __restrict__ out) {
  __shared__ float s1[256], s2[256];
  int row = blockIdx.x, tid = threadIdx.x;
  int b = row >> 10;
  const float* xr = x + (size_t)row * Dm;
  float v[3], sum = 0.f, sq = 0.f;
#pragma unroll
  for (int e = 0; e < 3; ++e) { v[e] = xr[tid + 256 * e]; sum += v[e]; sq += v[e] * v[e]; }
  s1[tid] = sum; s2[tid] = sq; __syncthreads();
  for (int off = 128; off > 0; off >>= 1) {
    if (tid < off) { s1[tid] += s1[tid + off]; s2[tid] += s2[tid + off]; }
    __syncthreads();
  }
  float mu = s1[0] * (1.0f / Dm);
  float var = s2[0] * (1.0f / Dm) - mu * mu;
  float rstd = rsqrtf(var + 1e-5f);
  const float* cb = cm + (size_t)b * cstride;
#pragma unroll
  for (int e = 0; e < 3; ++e) {
    int j = tid + 256 * e;
    float y = (v[e] - mu) * rstd * w[j];
    y = y * (1.0f + cb[sc_off + j]) + cb[sh_off + j];
    out[(size_t)row * Dm + j] = f2bf(y);
  }
}

// --------------------------- WMMA GEMM (64x32 per wave) ---------------------
// Y[R,N] = A_bf16[R,K] @ W_f32[N,K]^T  (+ epilogue per mode)
// mode 0: outF = acc                                   (qkv)
// mode 1: resid += cm[b][gate_off+col] * acc           (attn out proj)
// mode 2: outB  = bf16(gelu(acc + bias[col]))          (mlp fc1)
// mode 3: resid += cm[b][gate_off+col] * (acc+bias)    (mlp fc2)
// mode 4: outF  = acc + bias[col]                      (lm head)
__global__ void __launch_bounds__(256)
k_gemm(const __bf16* __restrict__ A, const float* __restrict__ W,
       const float* __restrict__ bias, const float* __restrict__ cm, int gate_off,
       float* __restrict__ resid, float* __restrict__ outF, __bf16* __restrict__ outB,
       int N, int K, int mode) {
  int lane = threadIdx.x & 31;
  int gw = blockIdx.x * 8 + (threadIdx.x >> 5);
  int ngroups = N >> 5;                              // N/32 col groups
  int gm = gw / ngroups, gn = gw - gm * ngroups;
  if (gm >= (Rn >> 6)) return;                       // wave-uniform
  int m0 = gm * 64, n0 = gn * 32;
  int nl = lane & 15, hh = lane >> 4;
  const __bf16* ap = A + (size_t)(m0 + nl) * K + hh * 16;
  const float*  wp = W + (size_t)(n0 + nl) * K + hh * 16;
  v8f acc[4][2];
#pragma unroll
  for (int mi = 0; mi < 4; ++mi)
#pragma unroll
    for (int ni = 0; ni < 2; ++ni) acc[mi][ni] = (v8f){};
  for (int k0 = 0; k0 < K; k0 += 32) {
    v16bf a[4], bw[2];
#pragma unroll
    for (int mi = 0; mi < 4; ++mi) {
      const __bf16* p = ap + (size_t)mi * 16 * K + k0;
      bf8v lo = *(const bf8v*)(p);
      bf8v hi = *(const bf8v*)(p + 8);
      a[mi] = SHUF16(lo, hi);
    }
#pragma unroll
    for (int ni = 0; ni < 2; ++ni) {
      const float* wb = wp + (size_t)ni * 16 * K + k0;
#pragma unroll
      for (int q = 0; q < 4; ++q) {
        f4v f = *(const f4v*)(wb + q * 4);
        bw[ni][4 * q + 0] = f2bf(f[0]);
        bw[ni][4 * q + 1] = f2bf(f[1]);
        bw[ni][4 * q + 2] = f2bf(f[2]);
        bw[ni][4 * q + 3] = f2bf(f[3]);
      }
    }
#pragma unroll
    for (int mi = 0; mi < 4; ++mi)
#pragma unroll
      for (int ni = 0; ni < 2; ++ni)
        acc[mi][ni] = __builtin_amdgcn_wmma_f32_16x16x32_bf16(
            false, a[mi], false, bw[ni], (short)0, acc[mi][ni], false, false);
  }
#pragma unroll
  for (int mi = 0; mi < 4; ++mi) {
#pragma unroll
    for (int ni = 0; ni < 2; ++ni) {
      int col = n0 + ni * 16 + nl;
#pragma unroll
      for (int v = 0; v < 8; ++v) {
        int r = m0 + mi * 16 + v + 8 * hh;
        float val = acc[mi][ni][v];
        if (mode == 0) {
          outF[(size_t)r * N + col] = val;
        } else if (mode == 1) {
          int b = r >> 10;
          resid[(size_t)r * Dm + col] += cm[(size_t)b * 6 * Dm + gate_off + col] * val;
        } else if (mode == 2) {
          outB[(size_t)r * N + col] = f2bf(geluf(val + bias[col]));
        } else if (mode == 3) {
          int b = r >> 10;
          resid[(size_t)r * Dm + col] +=
              cm[(size_t)b * 6 * Dm + gate_off + col] * (val + bias[col]);
        } else { // 4
          outF[(size_t)r * N + col] = val + bias[col];
        }
      }
    }
  }
}

// --------------------------- RoPE + layout split ----------------------------
// qkv f32 [b][s][3][H][HD] -> Q,K bf16 [b][h][s][d] (rotated), Vt bf16 [b][h][d][s]
__global__ void k_rope(const float* __restrict__ qkv, __bf16* __restrict__ Q,
                       __bf16* __restrict__ Kb, __bf16* __restrict__ Vt) {
  int i = blockIdx.x * 256 + threadIdx.x;            // Bn*Hn*Sn*HDn threads
  int d = i & 63, s = (i >> 6) & 1023, bh = i >> 16;
  int b = bh / Hn, h = bh - b * Hn;
  const float* row = qkv + ((size_t)b * Sn + s) * (3 * Dm);
  int dd = d & 31;
  float f = __expf(-LOG1E4 * (float)dd / 32.0f);     // inv_freq
  float cv = cosf((float)s * f), sv = sinf((float)s * f);
  int base = h * HDn + d;
  float qv = row[base];
  float qr = (d < 32) ? -row[base + 32] : row[base - 32];
  Q[((size_t)bh * Sn + s) * HDn + d] = f2bf(qv * cv + qr * sv);
  float kv = row[Dm + base];
  float kr = (d < 32) ? -row[Dm + base + 32] : row[Dm + base - 32];
  Kb[((size_t)bh * Sn + s) * HDn + d] = f2bf(kv * cv + kr * sv);
  Vt[((size_t)bh * HDn + d) * Sn + s] = f2bf(row[2 * Dm + base]);
}

// --------------------------- flash attention (WMMA) -------------------------
// One wave: 16 query rows of one (b,h); loops key tiles of 32.
__global__ void __launch_bounds__(256)
k_attn(const __bf16* __restrict__ Q, const __bf16* __restrict__ Kb,
       const __bf16* __restrict__ Vt, __bf16* __restrict__ O) {
  __shared__ __bf16 plds[8][16 * 32];
  int lane = threadIdx.x & 31, w = threadIdx.x >> 5;
  int gw = blockIdx.x * 8 + w;                       // Bn*Hn*(Sn/16) waves
  int bh = gw >> 6, r0 = (gw & 63) << 4;
  int b = bh / Hn, h = bh - b * Hn;
  int nl = lane & 15, hh = lane >> 4;

  // Q A-fragments (d=0..31 and d=32..63), resident
  const __bf16* qp = Q + ((size_t)bh * Sn + r0 + nl) * HDn + hh * 16;
  bf8v q0 = *(const bf8v*)(qp);
  bf8v q1 = *(const bf8v*)(qp + 8);
  bf8v q2 = *(const bf8v*)(qp + 32);
  bf8v q3 = *(const bf8v*)(qp + 40);
  v16bf a0 = SHUF16(q0, q1), a1 = SHUF16(q2, q3);

  v8f o0 = {}, o1 = {}, o2 = {}, o3 = {};
  float mrun[8], lrun[8];
#pragma unroll
  for (int v = 0; v < 8; ++v) { mrun[v] = -1e30f; lrun[v] = 0.f; }
  const float isq = 0.125f;                          // 1/sqrt(64)
  const __bf16* kbase = Kb + (size_t)bh * Sn * HDn;
  const __bf16* vbase = Vt + (size_t)bh * HDn * Sn;
  __bf16* pw = plds[w];

  for (int j0 = 0; j0 < Sn; j0 += 32) {
    v8f s0 = {}, s1 = {};
    {
      const __bf16* kr = kbase + (size_t)(j0 + nl) * HDn + hh * 16;
      v16bf b0 = *(const v16bf*)(kr);
      v16bf b1 = *(const v16bf*)(kr + 32);
      s0 = __builtin_amdgcn_wmma_f32_16x16x32_bf16(false, a0, false, b0, (short)0, s0, false, false);
      s0 = __builtin_amdgcn_wmma_f32_16x16x32_bf16(false, a1, false, b1, (short)0, s0, false, false);
      kr += 16 * HDn;
      b0 = *(const v16bf*)(kr);
      b1 = *(const v16bf*)(kr + 32);
      s1 = __builtin_amdgcn_wmma_f32_16x16x32_bf16(false, a0, false, b0, (short)0, s1, false, false);
      s1 = __builtin_amdgcn_wmma_f32_16x16x32_bf16(false, a1, false, b1, (short)0, s1, false, false);
    }
    // online softmax: C layout rows (v+8*hh), cols across 16 lanes per half;
    // xor-shuffle reductions (masks 1..8) stay inside each half.
#pragma unroll
    for (int v = 0; v < 8; ++v) {
      float x0 = s0[v] * isq, x1 = s1[v] * isq;
      float mx = fmaxf(x0, x1);
      for (int msk = 8; msk >= 1; msk >>= 1) mx = fmaxf(mx, __shfl_xor(mx, msk, 32));
      float mnew = fmaxf(mrun[v], mx);
      float p0 = __expf(x0 - mnew), p1 = __expf(x1 - mnew);
      float rs = p0 + p1;
      for (int msk = 8; msk >= 1; msk >>= 1) rs += __shfl_xor(rs, msk, 32);
      float corr = __expf(mrun[v] - mnew);
      lrun[v] = lrun[v] * corr + rs;
      mrun[v] = mnew;
      o0[v] *= corr; o1[v] *= corr; o2[v] *= corr; o3[v] *= corr;
      int rrow = (v + 8 * hh) * 32;
      pw[rrow + nl]      = f2bf(p0);
      pw[rrow + 16 + nl] = f2bf(p1);
    }
    asm volatile("s_wait_dscnt 0" ::: "memory");     // LDS P visible to all lanes
    v16bf pa;
    {
      const __bf16* pp = pw + nl * 32 + hh * 16;
      bf8v lo = *(const bf8v*)(pp);
      bf8v hi = *(const bf8v*)(pp + 8);
      pa = SHUF16(lo, hi);
    }
    // P(16x32) @ V(32x64): 4 d-tiles, B-frag = 16 contiguous j from Vt rows
    {
      const __bf16* vr = vbase + (size_t)nl * Sn + j0 + hh * 16;
      v16bf bv;
      bv = *(const v16bf*)(vr);
      o0 = __builtin_amdgcn_wmma_f32_16x16x32_bf16(false, pa, false, bv, (short)0, o0, false, false);
      bv = *(const v16bf*)(vr + 16 * Sn);
      o1 = __builtin_amdgcn_wmma_f32_16x16x32_bf16(false, pa, false, bv, (short)0, o1, false, false);
      bv = *(const v16bf*)(vr + 32 * Sn);
      o2 = __builtin_amdgcn_wmma_f32_16x16x32_bf16(false, pa, false, bv, (short)0, o2, false, false);
      bv = *(const v16bf*)(vr + 48 * Sn);
      o3 = __builtin_amdgcn_wmma_f32_16x16x32_bf16(false, pa, false, bv, (short)0, o3, false, false);
    }
  }
  __bf16* ob = O + ((size_t)b * Sn + r0) * Dm + h * HDn;
#pragma unroll
  for (int v = 0; v < 8; ++v) {
    float li = 1.0f / lrun[v];
    size_t ro = (size_t)(v + 8 * hh) * Dm;
    ob[ro + 0  + nl] = f2bf(o0[v] * li);
    ob[ro + 16 + nl] = f2bf(o1[v] * li);
    ob[ro + 32 + nl] = f2bf(o2[v] * li);
    ob[ro + 48 + nl] = f2bf(o3[v] * li);
  }
}

// --------------------------- host orchestration -----------------------------

extern "C" void kernel_launch(void* const* d_in, const int* in_sizes, int n_in,
                              void* d_out, int out_size, void* d_ws, size_t ws_size,
                              hipStream_t stream) {
  const int*   indices  = (const int*)  d_in[0];
  const float* sigma    = (const float*)d_in[1];
  const float* embedding= (const float*)d_in[2];
  const float* tmlp_w1  = (const float*)d_in[3];
  const float* tmlp_b1  = (const float*)d_in[4];
  const float* tmlp_w2  = (const float*)d_in[5];
  const float* tmlp_b2  = (const float*)d_in[6];
  const float* norm1_w  = (const float*)d_in[7];
  const float* qkv_w    = (const float*)d_in[8];
  const float* attn_o_w = (const float*)d_in[9];
  const float* norm2_w  = (const float*)d_in[10];
  const float* mlp_w1   = (const float*)d_in[11];
  const float* mlp_b1   = (const float*)d_in[12];
  const float* mlp_w2   = (const float*)d_in[13];
  const float* mlp_b2   = (const float*)d_in[14];
  const float* adaLN_w  = (const float*)d_in[15];
  const float* adaLN_b  = (const float*)d_in[16];
  const float* normf_w  = (const float*)d_in[17];
  const float* lin_w    = (const float*)d_in[18];
  const float* lin_b    = (const float*)d_in[19];
  const float* adaLNf_w = (const float*)d_in[20];
  const float* adaLNf_b = (const float*)d_in[21];
  float* out = (float*)d_out;

  char* ws = (char*)d_ws;
  size_t off = 0;
  auto alloc = [&](size_t bytes) { char* p = ws + off; off += (bytes + 255) & ~(size_t)255; return p; };
  float*  x    = (float*) alloc((size_t)Rn * Dm * 4);        // activations f32
  __bf16* xm   = (__bf16*)alloc((size_t)Rn * Dm * 2);        // modulated LN (bf16)
  float*  qkv  = (float*) alloc((size_t)Rn * 3 * Dm * 4);
  __bf16* Qb   = (__bf16*)alloc((size_t)Bn * Hn * Sn * HDn * 2);
  __bf16* Kb   = (__bf16*)alloc((size_t)Bn * Hn * Sn * HDn * 2);
  __bf16* Vt   = (__bf16*)alloc((size_t)Bn * Hn * HDn * Sn * 2);
  __bf16* Ob   = (__bf16*)alloc((size_t)Rn * Dm * 2);
  __bf16* Hb   = (__bf16*)alloc((size_t)Rn * MLPn * 2);
  float*  c6   = (float*) alloc((size_t)Ln * Bn * 6 * Dm * 4);
  float*  c2   = (float*) alloc((size_t)Bn * 2 * Dm * 4);
  float*  t1   = (float*) alloc((size_t)Bn * CONDn * 4);
  float*  cbuf = (float*) alloc((size_t)Bn * CONDn * 4);
  (void)ws_size;

  // conditioning
  k_embed_gather<<<(Rn * Dm) / 256, 256, 0, stream>>>(indices, embedding, x);
  k_t1<<<Bn, CONDn, 0, stream>>>(sigma, tmlp_w1, tmlp_b1, t1);
  k_t2<<<Bn, CONDn, 0, stream>>>(t1, tmlp_w2, tmlp_b2, cbuf);
  k_c6<<<(Ln * Bn * 6 * Dm) / 256, 256, 0, stream>>>(cbuf, adaLN_w, adaLN_b, c6);
  k_c2<<<(Bn * 2 * Dm) / 256, 256, 0, stream>>>(cbuf, adaLNf_w, adaLNf_b, c2);

  for (int l = 0; l < Ln; ++l) {
    const float* c6l = c6 + (size_t)l * Bn * 6 * Dm;
    // norm1 + modulate (sh_msa@0, sc_msa@768)
    k_ln_mod<<<Rn, 256, 0, stream>>>(x, norm1_w + (size_t)l * Dm, c6l, 6 * Dm, 0, Dm, xm);
    // qkv = xm @ qkv_w^T
    k_gemm<<<(Rn / 64) * (3 * Dm / 32) / 8, 256, 0, stream>>>(
        xm, qkv_w + (size_t)l * 3 * Dm * Dm, nullptr, nullptr, 0, nullptr, qkv, nullptr,
        3 * Dm, Dm, 0);
    // rope + split
    k_rope<<<(Bn * Hn * Sn * HDn) / 256, 256, 0, stream>>>(qkv, Qb, Kb, Vt);
    // attention
    k_attn<<<(Bn * Hn * (Sn / 16)) / 8, 256, 0, stream>>>(Qb, Kb, Vt, Ob);
    // x += g_msa * (o @ attn_out_w^T)   (g_msa @ offset 2*Dm)
    k_gemm<<<(Rn / 64) * (Dm / 32) / 8, 256, 0, stream>>>(
        Ob, attn_o_w + (size_t)l * Dm * Dm, nullptr, c6l, 2 * Dm, x, nullptr, nullptr,
        Dm, Dm, 1);
    // norm2 + modulate (sh_mlp@3*Dm, sc_mlp@4*Dm)
    k_ln_mod<<<Rn, 256, 0, stream>>>(x, norm2_w + (size_t)l * Dm, c6l, 6 * Dm, 3 * Dm, 4 * Dm, xm);
    // h = gelu(xm @ w1^T + b1)
    k_gemm<<<(Rn / 64) * (MLPn / 32) / 8, 256, 0, stream>>>(
        xm, mlp_w1 + (size_t)l * MLPn * Dm, mlp_b1 + (size_t)l * MLPn, nullptr, 0,
        nullptr, nullptr, Hb, MLPn, Dm, 2);
    // x += g_mlp * (h @ w2^T + b2)      (g_mlp @ offset 5*Dm)
    k_gemm<<<(Rn / 64) * (Dm / 32) / 8, 256, 0, stream>>>(
        Hb, mlp_w2 + (size_t)l * Dm * MLPn, mlp_b2 + (size_t)l * Dm, c6l, 5 * Dm, x,
        nullptr, nullptr, Dm, MLPn, 3);
  }

  // final LN + modulate (c2: sh@0, sc@Dm), then logits
  k_ln_mod<<<Rn, 256, 0, stream>>>(x, normf_w, c2, 2 * Dm, 0, Dm, xm);
  k_gemm<<<(Rn / 64) * (Vn / 32) / 8, 256, 0, stream>>>(
      xm, lin_w, lin_b, nullptr, 0, nullptr, out, nullptr, Vn, Dm, 4);
  (void)in_sizes; (void)n_in; (void)out_size;
}
